// CondConceptSampler_10952166605310
// MI455X (gfx1250) — matrix operand order repaired and checked by reading, not compiled, not run on hardware
//
#include <hip/hip_runtime.h>

#define B_   8
#define G_   32
#define BW_  64
#define S_   64
#define CD_  256
#define N_   4096
#define EPS_ 1e-5f

#define NT    64   // n-tile (columns per async stage)
#define NSTG  (N_ / NT)
#define FSTR  68   // fp32 transposed-tile row stride (272 B, 16B-aligned)
#define XSTR  72   // bf16 tile / Wk row stride (144 B, 16B-aligned)

#define AS1 __attribute__((address_space(1)))
#define AS3 __attribute__((address_space(3)))

#if defined(__gfx1250__) && __has_builtin(__builtin_amdgcn_global_load_async_to_lds_b32)
#define USE_ASYNC 1
#else
#define USE_ASYNC 0
#endif

#if USE_ASYNC
#if __has_builtin(__builtin_amdgcn_s_wait_asynccnt)
#define WAIT_ASYNC() __builtin_amdgcn_s_wait_asynccnt(0)
#else
#define WAIT_ASYNC() asm volatile("s_wait_asynccnt 0x0" ::: "memory")
#endif
#else
#define WAIT_ASYNC() do {} while (0)
#endif

typedef __attribute__((ext_vector_type(16))) __bf16 v16bf;
typedef __attribute__((ext_vector_type(8)))  float  v8f;

union V16U { uint4 q[2]; v16bf v; };

// round-to-nearest (half-up) f32 -> bf16; 2 VALU ops
__device__ __forceinline__ unsigned short f2bf(float f) {
  return (unsigned short)((__float_as_uint(f) + 0x8000u) >> 16);
}

// pack two f32 -> bf16x2 in 3 VALU ops: two rounding adds + one v_perm_b32
// (byte-select the high halves; a lands in [15:0], b in [31:16])
__device__ __forceinline__ unsigned int pk2bf(float a, float b) {
  unsigned int ua = __float_as_uint(a) + 0x8000u;
  unsigned int ub = __float_as_uint(b) + 0x8000u;
  return __builtin_amdgcn_perm(ub /*hi*/, ua /*lo*/, 0x07060302u);
}

// Stage one 64c x 64n fp32 tile, transposing on the fly: global reads are
// n-contiguous across lanes (coalesced), LDS destinations scatter per-lane
// into the transposed [n][c] layout (the async DMA does the transpose).
__device__ __forceinline__ void stage_tile(const float* __restrict__ xg,
                                           float* dstbase, int nb, int tid) {
  const int n  = tid & 63;
  const int c0 = (tid >> 6) << 4;          // 0,16,32,48
  float* drow = dstbase + n * FSTR;
  const float* gsrc = xg + nb + n;
  #pragma unroll
  for (int i = 0; i < 16; ++i) {
    const int c = c0 + i;
#if USE_ASYNC
    __builtin_amdgcn_global_load_async_to_lds_b32(
        (AS1 int*)(gsrc + (size_t)c * N_), (AS3 int*)(drow + c), 0, 0);
#else
    drow[c] = gsrc[(size_t)c * N_];
#endif
  }
}

__global__ __launch_bounds__(256)
void fused_gn_attn(const float* __restrict__ x,    const float* __restrict__ sent,
                   const float* __restrict__ Wq,   const float* __restrict__ Wk,
                   const float* __restrict__ Wv,   const float* __restrict__ g1,
                   const float* __restrict__ b1,   const float* __restrict__ g2,
                   const float* __restrict__ b2,   float* __restrict__ out) {
  __shared__ __align__(16) float          sXF[2][NT * FSTR]; // 34816 B, async dbl-buffer
  __shared__ __align__(16) unsigned short sPool[S_ * XSTR];  //  9216 B: Wk, then bf16 xT
  __shared__ float sSim[N_];                                 // 16384 B
  __shared__ float sQg[S_];
  __shared__ float sQr[S_];
  __shared__ float sOut[BW_];
  __shared__ float sRed[32];

  const int tid  = threadIdx.x;
  const int w    = tid >> 5;
  const int lane = tid & 31;
  const int bg   = blockIdx.x;
  const int b    = bg / G_, g = bg % G_;

  const float* xg = x + (size_t)(b * (G_ * BW_) + g * BW_) * N_;  // [64][4096]

  // kick off stage 0 immediately (overlaps with the q/Wk prologue)
  stage_tile(xg, sXF[0], 0, tid);

  // ---------- q = Wq[g] @ sent[b]; GroupNorm over S; qg = qn * gamma2 ----------
  {
    int s = tid >> 2, p = tid & 3;
    const float* wq = Wq + (size_t)(g * S_ + s) * CD_ + p * 64;
    const float* se = sent + (size_t)b * CD_ + p * 64;
    float acc = 0.f;
    #pragma unroll 8
    for (int c = 0; c < 64; ++c) acc += wq[c] * se[c];
    sSim[tid] = acc;                       // scratch
  }
  __syncthreads();
  if (tid < S_)
    sQr[tid] = sSim[tid*4] + sSim[tid*4+1] + sSim[tid*4+2] + sSim[tid*4+3];
  __syncthreads();
  if (tid == 0) {
    float m = 0.f, v = 0.f;
    for (int s = 0; s < S_; ++s) m += sQr[s];
    m *= (1.0f / S_);
    for (int s = 0; s < S_; ++s) { float d = sQr[s] - m; v += d * d; }
    v *= (1.0f / S_);
    sRed[0] = m; sRed[1] = rsqrtf(v + EPS_);
  }
  __syncthreads();
  if (tid < S_) {
    float qn = (sQr[tid] - sRed[0]) * sRed[1] * g1[g*S_ + tid] + b1[g*S_ + tid];
    sQg[tid] = qn * g2[g*S_ + tid];
  }

  // ---------- stage Wk[g] as bf16 into sPool; zero sim0 accumulator ----------
  for (int i = 0; i < 16; ++i) {
    int e = tid + 256 * i;                 // 4096 elements
    int s = e >> 6, c = e & 63;
    sPool[s*XSTR + c] = f2bf(Wk[(size_t)(g*S_ + s) * BW_ + c]);
    sSim[e] = 0.f;
  }
  __syncthreads();

  // ---------- hoisted A operands: this wave's fixed Wk s-tile (K=0..63) ----------
  const int st    = w & 3;                       // s-tile 0..3 (2 waves each)
  const int srow  = st*16 + (lane & 15);         // A: M = lane%16 (both halves)
  const int chalf = (lane >> 4) * 8;             // lane half selects K sub-range
  V16U A0, A1;
  {
    const char* pr = (const char*)(sPool + srow * XSTR);
    A0.q[0] = *(const uint4*)(pr + chalf*2);           // K =  chalf..+7
    A0.q[1] = *(const uint4*)(pr + chalf*2 + 32);      // K = 16+chalf..+7
    A1.q[0] = *(const uint4*)(pr + 64 + chalf*2);      // K = 32+chalf..+7
    A1.q[1] = *(const uint4*)(pr + 64 + chalf*2 + 32); // K = 48+chalf..+7
  }
  const int rowoff = st*16 + ((lane >> 4) << 3);  // D: VGPR r -> row r / r+8
  float qrw[8];
  #pragma unroll
  for (int r = 0; r < 8; ++r) qrw[r] = sQg[rowoff + r];

  // ---------- main loop: async-staged k-GEMM; accumulate sum, sumsq, qg.k ----------
  float ksum = 0.f, ksq = 0.f;
  for (int nt = 0; nt < NSTG; ++nt) {
    const int nb  = nt * NT;
    const int buf = nt & 1;
    WAIT_ASYNC();                     // my stage-nt b32 asyncs have landed
    __syncthreads();                  // everyone's landed; A-hoist / prev compute done
    if (nt + 1 < NSTG)                // stream next stage into other buffer
      stage_tile(xg, sXF[buf ^ 1], nb + NT, tid);

    // convert once per element: fp32 transposed tile -> bf16 transposed tile
    {
      const int n  = tid & 63;
      const int c0 = (tid >> 6) << 4;              // 0,16,32,48
      const float* fr = sXF[buf] + n * FSTR + c0;
      float4 a0 = *(const float4*)(fr);
      float4 a1 = *(const float4*)(fr + 4);
      float4 a2 = *(const float4*)(fr + 8);
      float4 a3 = *(const float4*)(fr + 12);
      uint4 p0 = make_uint4(pk2bf(a0.x,a0.y), pk2bf(a0.z,a0.w),
                            pk2bf(a1.x,a1.y), pk2bf(a1.z,a1.w));
      uint4 p1 = make_uint4(pk2bf(a2.x,a2.y), pk2bf(a2.z,a2.w),
                            pk2bf(a3.x,a3.y), pk2bf(a3.z,a3.w));
      uint4* dst = (uint4*)((char*)(sPool) + (n * XSTR + c0) * 2);
      dst[0] = p0;
      dst[1] = p1;
    }
    __syncthreads();                  // bf16 tile ready

    #pragma unroll
    for (int j = 0; j < 2; ++j) {
      const int nsub = (w >> 2) + 2 * j;            // 0..3
      const int ncol = nsub*16 + (lane & 15);       // B: N = lane%16
      const int kh   = (lane >> 4) * 8;             // B: K sub-range per lane group
      const char* bp = (const char*)(sPool + ncol * XSTR);
      V16U B0, B1;
      B0.q[0] = *(const uint4*)(bp + kh*2);         // K = kh..kh+7
      B0.q[1] = *(const uint4*)(bp + kh*2 + 32);    // K = 16+kh..16+kh+7
      B1.q[0] = *(const uint4*)(bp + 64 + kh*2);    // K = 32+kh..
      B1.q[1] = *(const uint4*)(bp + 64 + kh*2 + 32);
      v8f acc = {};
      acc = __builtin_amdgcn_wmma_f32_16x16x32_bf16(false, A0.v, false, B0.v,
                                                    (short)0, acc, false, false);
      acc = __builtin_amdgcn_wmma_f32_16x16x32_bf16(false, A1.v, false, B1.v,
                                                    (short)0, acc, false, false);
      float v0 = acc[0];
      float lsum = v0, lsq = v0 * v0, lsim = qrw[0] * v0;
      #pragma unroll
      for (int r = 1; r < 8; ++r) {
        float v = acc[r];
        lsum += v;
        lsq  = fmaf(v, v, lsq);
        lsim = fmaf(qrw[r], v, lsim);
      }
      ksum += lsum;
      ksq  += lsq;
      atomicAdd(&sSim[nb + ncol], lsim);            // ds_add_f32
    }
    __syncthreads();                  // tile fully consumed before next convert
  }
  __syncthreads();

  // ---------- mu / var over raw k -> rstd ----------
  for (int off = 16; off > 0; off >>= 1) {
    ksum += __shfl_down(ksum, off, 32);
    ksq  += __shfl_down(ksq,  off, 32);
  }
  if (lane == 0) { sRed[w] = ksum; sRed[8 + w] = ksq; }
  __syncthreads();
  if (tid == 0) {
    float ts = 0.f, tq = 0.f;
    for (int i = 0; i < 8; ++i) { ts += sRed[i]; tq += sRed[8 + i]; }
    const float invn = 1.0f / ((float)S_ * (float)N_);
    float mu  = ts * invn;
    float var = tq * invn - mu * mu;
    sRed[16] = rsqrtf(var + EPS_);
  }
  __syncthreads();
  const float rstd = sRed[16];

  // ---------- softmax over logits = rstd * sim0 (additive consts cancel) ----------
  float lmax = -3.0e38f;
  for (int i = 0; i < 16; ++i) lmax = fmaxf(lmax, sSim[tid + 256*i]);
  for (int off = 16; off > 0; off >>= 1) lmax = fmaxf(lmax, __shfl_down(lmax, off, 32));
  if (lane == 0) sRed[w] = lmax;
  __syncthreads();
  if (tid == 0) {
    float m = sRed[0];
    for (int i = 1; i < 8; ++i) m = fmaxf(m, sRed[i]);
    sRed[17] = m * rstd;
  }
  __syncthreads();
  const float gmax = sRed[17];
  float esum = 0.f;
  for (int i = 0; i < 16; ++i) {
    int idx = tid + 256 * i;
    float e = __expf(rstd * sSim[idx] - gmax);
    sSim[idx] = e;
    esum += e;
  }
  for (int off = 16; off > 0; off >>= 1) esum += __shfl_down(esum, off, 32);
  if (lane == 0) sRed[w] = esum;
  __syncthreads();
  if (tid == 0) {
    float t = 0.f;
    for (int i = 0; i < 8; ++i) t += sRed[i];
    sRed[18] = 1.0f / t;
  }
  __syncthreads();
  const float inv = sRed[18];

  // ---------- value pass (fp32 exact, L2-hot): out[c] = inv * sum_n e[n]*x[c,n] ----------
  for (int i = 0; i < 8; ++i) {
    const int c = w + 8 * i;
    const float* xr = xg + (size_t)c * N_;
    float acc = 0.f;
    for (int k2 = 0; k2 < N_ / 128; ++k2) {
      int n0 = (k2 * 32 + lane) * 4;
      float4 xv = *(const float4*)(xr + n0);
      acc += sSim[n0] * xv.x + sSim[n0+1] * xv.y + sSim[n0+2] * xv.z + sSim[n0+3] * xv.w;
    }
    for (int off = 16; off > 0; off >>= 1) acc += __shfl_down(acc, off, 32);
    if (lane == 0) sOut[c] = acc * inv;
  }
  __syncthreads();

  // ---------- final grouped conv: res[s] = Wv[g,s,:] . out ----------
  if (tid < S_) {
    const float* wv = Wv + (size_t)(g * S_ + tid) * BW_;
    float r = 0.f;
    #pragma unroll 8
    for (int c = 0; c < BW_; ++c) r += wv[c] * sOut[c];
    out[(size_t)bg * S_ + tid] = r;
  }
}

extern "C" void kernel_launch(void* const* d_in, const int* in_sizes, int n_in,
                              void* d_out, int out_size, void* d_ws, size_t ws_size,
                              hipStream_t stream) {
  const float* x    = (const float*)d_in[0];
  const float* sent = (const float*)d_in[1];
  const float* Wq   = (const float*)d_in[2];
  const float* Wk   = (const float*)d_in[3];
  const float* Wv   = (const float*)d_in[4];
  const float* g1   = (const float*)d_in[5];
  const float* b1   = (const float*)d_in[6];
  const float* g2   = (const float*)d_in[7];
  const float* b2   = (const float*)d_in[8];
  float* out = (float*)d_out;
  (void)in_sizes; (void)n_in; (void)out_size; (void)d_ws; (void)ws_size;
  fused_gn_attn<<<B_ * G_, 256, 0, stream>>>(x, sent, Wq, Wk, Wv, g1, b1, g2, b2, out);
}
